// FlaxMixtralAttention_12206297055486
// MI455X (gfx1250) — matrix operand order repaired and checked by reading, not compiled, not run on hardware
//
#include <hip/hip_runtime.h>
#include <hip/hip_bf16.h>

// ---------------------------------------------------------------------------
// MI455X (gfx1250): bf16 WMMA everywhere, fp32 accumulate.
//   B=2, S=2048, HID=4096, H=32, KH=8, D=128, WINDOW=1024, THETA=1e6
// Workspace (bf16 raw u16): q[4096,4096] k[4096,1024] vT[1024,4096]
//                           ao[4096,4096]  -> ~84 MB total
// CDNA5-specific paths used:
//   * v_wmma_f32_16x16x32_bf16 for all matmuls
//   * global_load_async_to_lds_b128 + s_wait_asynccnt for K/V tile staging
// V is produced TRANSPOSED (vT[n][token]) by its projection GEMM so that the
// attention P*V B-operand (keys contiguous per lane) loads as ds_load_b128.
// ---------------------------------------------------------------------------

typedef __bf16 v16bf __attribute__((ext_vector_type(16)));
typedef float  v8f   __attribute__((ext_vector_type(8)));

union Frag {
  v16bf v;
  unsigned int w[8];
  unsigned short u[16];
};

__device__ __forceinline__ unsigned short f2bf(float f) {
  unsigned int x = __float_as_uint(f);
  x += 0x7FFFu + ((x >> 16) & 1u);          // round-to-nearest-even
  return (unsigned short)(x >> 16);
}
__device__ __forceinline__ float bf2f(unsigned short h) {
  return __uint_as_float(((unsigned int)h) << 16);
}

__device__ __forceinline__ v8f wmma_bf16(const Frag& a, const Frag& b, v8f c) {
  // D(16x16 f32) = A(16x32 bf16) * B(32x16 bf16) + C
  return __builtin_amdgcn_wmma_f32_16x16x32_bf16(
      false, a.v, false, b.v, (short)0, c, false, false);
}

// Fragment K-index maps used here (contiguous k, same for A and B; not the ISA table order):
//  A 16x32 (16-bit): lane m = l&15; kA(v) = 2*v + (l>>4)*16                    (+1)
//  B 32x16 (16-bit): lane n = l&15; kB(v) = 2*v + (l>>4)*16                    (+1)
//  C 16x16 (f32):    row m = v + (l>>4)*8 ; col n = l&15

// 4 x 16-byte async copies global -> LDS (per lane); IOFFSET advances both
// the LDS and global addresses (ISA 08_async_tensor §4.4).
__device__ __forceinline__ void async_copy64B(unsigned lds_addr,
                                              unsigned long long gaddr) {
  asm volatile(
      "global_load_async_to_lds_b128 %0, %1, off\n\t"
      "global_load_async_to_lds_b128 %0, %1, off offset:16\n\t"
      "global_load_async_to_lds_b128 %0, %1, off offset:32\n\t"
      "global_load_async_to_lds_b128 %0, %1, off offset:48"
      :: "v"(lds_addr), "v"(gaddr) : "memory");
}
__device__ __forceinline__ void async_wait0() {
  asm volatile("s_wait_asynccnt 0x0" ::: "memory");
}

// ---------------------------------------------------------------------------
// WMMA GEMM: C[M,N] = A[M,K](row) * Bw[K,N](row, fp32)
// AT=float  -> convert A to bf16 while staging; AT=ushort -> bf16 raw copy.
// MODE: 0 = bf16 row-major out, 1 = fp32 row-major out,
//       2 = bf16 TRANSPOSED out (C^T[n][m], stride M) -- used for V.
// Block tile 64(M) x 128(N), BK=32, 256 threads = 8 waves (4 M x 2 N groups).
// B tile staged TRANSPOSED in LDS (sBt[n][k], 16B-aligned padded rows) so
// each B-fragment is two contiguous ds_load_b128.
// ---------------------------------------------------------------------------
template <typename AT, int MODE>
__global__ __launch_bounds__(256) void gemm_wmma(
    const AT* __restrict__ A, const float* __restrict__ Bw,
    void* __restrict__ Cp, int M, int N, int K)
{
  __shared__ unsigned short sA [64 * 40];   // [m][k]  pad 32->40
  __shared__ unsigned short sBt[128 * 40];  // [n][k]  pad 32->40 (80B rows)

  const int tid  = threadIdx.x;
  const int wid  = tid >> 5;
  const int lane = tid & 31;
  const int half = lane >> 4;
  const int l16  = lane & 15;
  const int wm   = wid & 3;                 // M sub-tile (16 rows)
  const int wn   = wid >> 2;                // N group (64 cols)
  const int bm   = blockIdx.y * 64;
  const int bn   = blockIdx.x * 128;

  const float Z = 0.0f;
  v8f acc[4] = { {Z,Z,Z,Z,Z,Z,Z,Z}, {Z,Z,Z,Z,Z,Z,Z,Z},
                 {Z,Z,Z,Z,Z,Z,Z,Z}, {Z,Z,Z,Z,Z,Z,Z,Z} };

  const int arow = tid >> 2, acol = (tid & 3) * 8;   // A: 64r x 32c, 8 each
  const int brow = tid >> 3, bcol = (tid & 7) * 16;  // B: 32k x 128n, 16 each

  for (int kb = 0; kb < K; kb += 32) {
    // ---- stage A tile [m][k] ----
    {
      const AT* src = A + (size_t)(bm + arow) * K + kb + acol;
      unsigned short* dst = &sA[arow * 40 + acol];
      if constexpr (sizeof(AT) == 4) {
        const float4 f0 = ((const float4*)src)[0];
        const float4 f1 = ((const float4*)src)[1];
        dst[0] = f2bf(f0.x); dst[1] = f2bf(f0.y); dst[2] = f2bf(f0.z); dst[3] = f2bf(f0.w);
        dst[4] = f2bf(f1.x); dst[5] = f2bf(f1.y); dst[6] = f2bf(f1.z); dst[7] = f2bf(f1.w);
      } else {
        *(uint4*)dst = *(const uint4*)src;
      }
    }
    // ---- stage B tile transposed [n][k] (fp32 -> bf16) ----
    {
      const float* src = Bw + (size_t)(kb + brow) * N + bn + bcol;
#pragma unroll
      for (int t = 0; t < 4; ++t) {
        const float4 f = ((const float4*)src)[t];
        sBt[(bcol + t * 4 + 0) * 40 + brow] = f2bf(f.x);
        sBt[(bcol + t * 4 + 1) * 40 + brow] = f2bf(f.y);
        sBt[(bcol + t * 4 + 2) * 40 + brow] = f2bf(f.z);
        sBt[(bcol + t * 4 + 3) * 40 + brow] = f2bf(f.w);
      }
      if (kb + 32 < K)  // warm L2/WGP$ for the next B panel
        __builtin_prefetch(Bw + (size_t)(kb + 32 + brow) * N + bn + bcol, 0, 1);
    }
    __syncthreads();

    // ---- compute: 1 A-frag, 4 B-frags, 4 WMMAs per wave ----
    Frag af;
#pragma unroll
    for (int v = 0; v < 8; ++v) {
      const int k = v * 2 + half * 16;
      af.w[v] = *(const unsigned int*)&sA[(wm * 16 + l16) * 40 + k];
    }
#pragma unroll
    for (int nt = 0; nt < 4; ++nt) {
      Frag bf;
      const unsigned short* bb =
          &sBt[(wn * 64 + nt * 16 + l16) * 40 + half * 16];
#pragma unroll
      for (int v = 0; v < 8; ++v)
        bf.w[v] = *(const unsigned int*)(bb + 2 * v);   // k = 2v (+half*16)
      acc[nt] = wmma_bf16(af, bf, acc[nt]);
    }
    __syncthreads();
  }

  // ---- epilogue ----
#pragma unroll
  for (int nt = 0; nt < 4; ++nt) {
#pragma unroll
    for (int v = 0; v < 8; ++v) {
      const int row = bm + wm * 16 + v + half * 8;
      const int col = bn + wn * 64 + nt * 16 + l16;
      if constexpr (MODE == 1)
        ((float*)Cp)[(size_t)row * N + col] = acc[nt][v];
      else if constexpr (MODE == 2)
        ((unsigned short*)Cp)[(size_t)col * M + row] = f2bf(acc[nt][v]);
      else
        ((unsigned short*)Cp)[(size_t)row * N + col] = f2bf(acc[nt][v]);
    }
  }
}

// ---------------------------------------------------------------------------
// RoPE in place on a bf16 [B*S, nH*128] buffer. One thread per rotation pair.
// ---------------------------------------------------------------------------
__global__ __launch_bounds__(256) void rope_kernel(
    unsigned short* __restrict__ qk, const int* __restrict__ pos,
    int nHeads, int total)
{
  const int idx = blockIdx.x * blockDim.x + threadIdx.x;
  if (idx >= total) return;
  const int i = idx & 63;                       // pair index 0..63
  const int h = (idx >> 6) % nHeads;
  const int t = idx / (64 * nHeads);            // token = b*S + s
  const size_t base = ((size_t)t * nHeads + h) * 128;

  const float x1 = bf2f(qk[base + i]);
  const float x2 = bf2f(qk[base + i + 64]);
  const float p  = (float)pos[t];
  // inv_freq = theta^(-2i/128), ln(1e6) = 13.815510557964274
  const float inv = __expf(-((float)(2 * i) * (1.0f / 128.0f)) * 13.815510557964274f);
  float sn, cs;
  __sincosf(p * inv, &sn, &cs);
  qk[base + i]      = f2bf(x1 * cs - x2 * sn);
  qk[base + i + 64] = f2bf(x2 * cs + x1 * sn);
}

// ---------------------------------------------------------------------------
// Flash attention with sliding-window causal mask.
// Grid: (S/64, H, B). 128 threads = 4 waves; wave w owns 16 query rows.
// K tile [32 keys][128 d] and V tile [128 d][32 keys] staged via
// global_load_async_to_lds_b128.  Since S, the 64-query tile, and the
// 32-key tile all align, key indices never exceed S (jb+31 <= S-1): no
// clamping needed; the causal/window mask handles validity.
// ---------------------------------------------------------------------------
#define LDK  136   // K tile row pad: 128 -> 136 shorts
#define LDVT 40    // V^T tile row pad: 32 -> 40 shorts (80B, 16B aligned)
#define LDP  40

__global__ __launch_bounds__(128) void attn_kernel(
    const unsigned short* __restrict__ Q,   // [B*S, H*D]
    const unsigned short* __restrict__ Kb,  // [B*S, KH*D]
    const unsigned short* __restrict__ Vt,  // [KH*D, B*S]  (transposed V)
    unsigned short* __restrict__ Ob)        // [B*S, H*D]
{
  const int S = 2048, D = 128, H = 32, KH = 8, WIN = 1024;

  __shared__ unsigned short ldsK [32 * LDK];    // [key][d]
  __shared__ unsigned short ldsVt[128 * LDVT];  // [d][key]
  __shared__ unsigned short ldsP [4][16 * LDP];

  const int qt  = blockIdx.x;
  const int h   = blockIdx.y;
  const int b   = blockIdx.z;
  const int kvh = h >> 2;                       // GQA: H/KH = 4
  const int tid = threadIdx.x;
  const int wid = tid >> 5;
  const int lane = tid & 31;
  const int half = lane >> 4;
  const int l16  = lane & 15;
  const int q0   = qt * 64;
  const int qrow = q0 + wid * 16;               // wave's first query row

  // staging slots: K = 64B of one key row; V = 64B of one d row (32 keys)
  const int kk  = tid >> 2;                     // key row in tile (0..31)
  const int seg = (tid & 3) * 32;               // 32-bf16 chunk of d
  const unsigned ldsKa  = (unsigned)(unsigned long long)&ldsK[kk * LDK + seg];
  const unsigned ldsVta = (unsigned)(unsigned long long)&ldsVt[tid * LDVT];
  const size_t kbase = ((size_t)(b * S + kk) * KH + kvh) * D + seg;
  const size_t vbase = ((size_t)(kvh * D + tid)) * (size_t)(2 * S) + (size_t)b * S;

  // ---- preload Q fragments (16 rows x 128) straight from global ----
  Frag qf[4];
  {
    const unsigned short* qb =
        Q + (((size_t)(b * S + qrow + l16)) * H + h) * D;
#pragma unroll
    for (int kc = 0; kc < 4; ++kc)
#pragma unroll
      for (int v = 0; v < 8; ++v) {
        const int kd = kc * 32 + v * 2 + half * 16;
        qf[kc].w[v] = *(const unsigned int*)(qb + kd);
      }
  }

  const float Z = 0.0f;
  v8f of[8];
#pragma unroll
  for (int f = 0; f < 8; ++f) of[f] = (v8f){Z,Z,Z,Z,Z,Z,Z,Z};
  float mi[8], li[8];
#pragma unroll
  for (int v = 0; v < 8; ++v) { mi[v] = -1e30f; li[v] = 0.0f; }

  const int hi = q0 + 63;
  int lo = q0 - (WIN - 1); if (lo < 0) lo = 0; lo &= ~31;

  for (int jb = lo; jb <= hi; jb += 32) {
    // ---- async stage K [32][128] and V^T [128][32] tiles ----
    async_copy64B(ldsKa,  (unsigned long long)(Kb + kbase + (size_t)jb * KH * D));
    async_copy64B(ldsVta, (unsigned long long)(Vt + vbase + jb));
    async_wait0();
    __syncthreads();

    // ---- scores S = Q K^T : 16 x 32 per wave (two 16x16 frags) ----
    v8f sc[2];
#pragma unroll
    for (int nh = 0; nh < 2; ++nh) {
      v8f c = (v8f){Z,Z,Z,Z,Z,Z,Z,Z};
#pragma unroll
      for (int kc = 0; kc < 4; ++kc) {
        Frag bk;
        const unsigned short* kb_ = &ldsK[(nh * 16 + l16) * LDK + kc * 32 + half * 16];
#pragma unroll
        for (int v = 0; v < 8; ++v)
          bk.w[v] = *(const unsigned int*)(kb_ + 2 * v);
        c = wmma_bf16(qf[kc], bk, c);
      }
      sc[nh] = c;
    }

    // ---- mask + online softmax (fp32) ----
    const float scl = 0.08838834764831845f;     // 1/sqrt(128)
    float tmax[8];
#pragma unroll
    for (int v = 0; v < 8; ++v) {
      const int irow = qrow + v + half * 8;
#pragma unroll
      for (int nh = 0; nh < 2; ++nh) {
        const int j = jb + nh * 16 + l16;
        const float s = sc[nh][v] * scl;
        const bool ok = (j <= irow) && (irow - j < WIN);
        sc[nh][v] = ok ? s : -1e30f;
      }
      tmax[v] = fmaxf(sc[0][v], sc[1][v]);
    }
#pragma unroll
    for (int m = 1; m < 16; m <<= 1)
#pragma unroll
      for (int v = 0; v < 8; ++v)
        tmax[v] = fmaxf(tmax[v], __shfl_xor(tmax[v], m, 32));

    float alpha[8], rs[8];
#pragma unroll
    for (int v = 0; v < 8; ++v) {
      const float mn = fmaxf(mi[v], tmax[v]);
      alpha[v] = __expf(mi[v] - mn);
      mi[v] = mn;
      sc[0][v] = __expf(sc[0][v] - mn);
      sc[1][v] = __expf(sc[1][v] - mn);
      rs[v] = sc[0][v] + sc[1][v];
    }
#pragma unroll
    for (int m = 1; m < 16; m <<= 1)
#pragma unroll
      for (int v = 0; v < 8; ++v)
        rs[v] += __shfl_xor(rs[v], m, 32);
#pragma unroll
    for (int v = 0; v < 8; ++v) li[v] = li[v] * alpha[v] + rs[v];
#pragma unroll
    for (int f = 0; f < 8; ++f)
#pragma unroll
      for (int v = 0; v < 8; ++v) of[f][v] *= alpha[v];

    // ---- P: C-layout -> LDS -> A-layout fragment (bf16) ----
    unsigned short* P = &ldsP[wid][0];
#pragma unroll
    for (int v = 0; v < 8; ++v) {
      const int m = v + half * 8;
      P[m * LDP + l16]      = f2bf(sc[0][v]);
      P[m * LDP + 16 + l16] = f2bf(sc[1][v]);
    }
    Frag pf;
#pragma unroll
    for (int v = 0; v < 8; ++v) {
      const int k = v * 2 + half * 16;
      pf.w[v] = *(const unsigned int*)&P[l16 * LDP + k];
    }

    // ---- O += P V : 8 WMMAs across D (V^T tile: keys contiguous) ----
#pragma unroll
    for (int f = 0; f < 8; ++f) {
      Frag bv;
      const unsigned short* vb_ = &ldsVt[(f * 16 + l16) * LDVT + half * 16];
#pragma unroll
      for (int v = 0; v < 8; ++v)
        bv.w[v] = *(const unsigned int*)(vb_ + 2 * v);  // k = 2v (+half*16)
      of[f] = wmma_bf16(pf, bv, of[f]);
    }
    __syncthreads();
  }

  // ---- epilogue: O / l -> bf16 attn-out [B*S, H*D] ----
#pragma unroll
  for (int f = 0; f < 8; ++f)
#pragma unroll
    for (int v = 0; v < 8; ++v) {
      const int irow = qrow + v + half * 8;
      const size_t o = (((size_t)(b * S + irow)) * H + h) * D + f * 16 + l16;
      Ob[o] = f2bf(of[f][v] / li[v]);
    }
}

// ---------------------------------------------------------------------------
extern "C" void kernel_launch(void* const* d_in, const int* in_sizes, int n_in,
                              void* d_out, int out_size, void* d_ws, size_t ws_size,
                              hipStream_t stream) {
  const float* hs  = (const float*)d_in[0];   // [B,S,HID] = [4096,4096]
  const int*   pos = (const int*)  d_in[1];   // [B,S]
  const float* wq  = (const float*)d_in[2];   // [4096,4096]
  const float* wk  = (const float*)d_in[3];   // [4096,1024]
  const float* wv  = (const float*)d_in[4];   // [4096,1024]
  const float* wo  = (const float*)d_in[5];   // [4096,4096]
  float* out = (float*)d_out;                 // [4096,4096] fp32

  unsigned short* qb  = (unsigned short*)d_ws;
  unsigned short* kb  = qb  + (size_t)4096 * 4096;
  unsigned short* vtb = kb  + (size_t)4096 * 1024;  // transposed V [1024,4096]
  unsigned short* ab  = vtb + (size_t)4096 * 1024;  // ~84 MB total

  const dim3 gblk(256);
  // QKV projections (fp32 in -> bf16 out); V written transposed (MODE 2)
  gemm_wmma<float, 0><<<dim3(4096 / 128, 4096 / 64), gblk, 0, stream>>>(
      hs, wq, qb, 4096, 4096, 4096);
  gemm_wmma<float, 0><<<dim3(1024 / 128, 4096 / 64), gblk, 0, stream>>>(
      hs, wk, kb, 4096, 1024, 4096);
  gemm_wmma<float, 2><<<dim3(1024 / 128, 4096 / 64), gblk, 0, stream>>>(
      hs, wv, vtb, 4096, 1024, 4096);

  // RoPE in place (V needs none)
  rope_kernel<<<(4096 * 32 * 64) / 256, 256, 0, stream>>>(qb, pos, 32, 4096 * 32 * 64);
  rope_kernel<<<(4096 * 8 * 64) / 256, 256, 0, stream>>>(kb, pos, 8, 4096 * 8 * 64);

  // Sliding-window flash attention
  attn_kernel<<<dim3(2048 / 64, 32, 2), dim3(128), 0, stream>>>(qb, kb, vtb, ab);

  // Output projection (bf16 in -> fp32 out)
  gemm_wmma<unsigned short, 1><<<dim3(4096 / 128, 4096 / 64), gblk, 0, stream>>>(
      ab, wo, out, 4096, 4096, 4096);
}
